// OrientedRCNNHead_35373350650326
// MI455X (gfx1250) — matrix operand order repaired and checked by reading, not compiled, not run on hardware
//
#include <hip/hip_runtime.h>
#include <math.h>

// ---------------- types ----------------
typedef __attribute__((ext_vector_type(16))) __bf16          v16bf;
typedef __attribute__((ext_vector_type(8)))  float           v8f;
typedef __attribute__((ext_vector_type(8)))  unsigned short  us8;
typedef __attribute__((ext_vector_type(16))) unsigned short  us16;

__device__ __forceinline__ unsigned short f2bf(float f) {
  unsigned u = __builtin_bit_cast(unsigned, f);
  u += 0x7FFFu + ((u >> 16) & 1u);        // round-to-nearest-even
  return (unsigned short)(u >> 16);
}
__device__ __forceinline__ float bf2f(unsigned short h) {
  unsigned u = ((unsigned)h) << 16;
  return __builtin_bit_cast(float, u);
}

// async copy: 16 bytes global -> LDS, tracked by ASYNCcnt (CDNA5)
__device__ __forceinline__ void async_ld16(unsigned lds_off, const void* gptr) {
  asm volatile("global_load_async_to_lds_b128 %0, %1, off"
               :: "v"(lds_off), "v"(gptr) : "memory");
}

#define NB 2
#define NN 1024
#define NROI (NB * NN)
#define CCH 256
#define OUT7 7
#define FIN 12544           // 256*49
#define HID 1024

// ---------------- 1. f32 -> bf16 convert ----------------
__global__ void k_cvt_bf16(const float* __restrict__ src,
                           unsigned short* __restrict__ dst, int n) {
  int i = blockIdx.x * blockDim.x + threadIdx.x;
  if (i < n) dst[i] = f2bf(src[i]);
}

// ---------------- 2. NCHW f32 -> NHWC bf16 ----------------
__global__ void k_nchw_to_nhwc(const float* __restrict__ src,
                               unsigned short* __restrict__ dst,
                               int Hn) {
  int i = blockIdx.x * blockDim.x + threadIdx.x;
  int total = NB * CCH * Hn * Hn;
  if (i >= total) return;
  int c = i % CCH; int t = i / CCH;
  int x = t % Hn;  t /= Hn;
  int y = t % Hn;  int b = t / Hn;
  dst[i] = f2bf(src[((size_t)(b * CCH + c) * Hn + y) * Hn + x]);
}

// ---------------- 3. encode_theta ----------------
__global__ void k_encode(const float* __restrict__ prop,
                         float* __restrict__ boxes5) {
  int r = blockIdx.x * blockDim.x + threadIdx.x;
  if (r >= NROI) return;
  const float* p = prop + (size_t)r * 8;
  float x0 = p[0], y0 = p[1], x1 = p[2], y1 = p[3];
  float x2 = p[4], y2 = p[5], x3 = p[6], y3 = p[7];
  float cx = 0.25f * (x0 + x1 + x2 + x3);
  float cy = 0.25f * (y0 + y1 + y2 + y3);
  float e1x = x1 - x0, e1y = y1 - y0, e2x = x3 - x0, e2y = y3 - y0;
  float w = sqrtf(e1x * e1x + e1y * e1y);
  float h = sqrtf(e2x * e2x + e2y * e2y);
  float th = atan2f(e1y, e1x);
  float* o = boxes5 + (size_t)r * 5;
  o[0] = cx; o[1] = cy; o[2] = w; o[3] = h; o[4] = th;
}

// ---------------- 4. rotated ROI align (NHWC bf16 in, X bf16 out) --------
__global__ __launch_bounds__(256)
void k_roialign(const unsigned short* __restrict__ f0,
                const unsigned short* __restrict__ f1,
                const unsigned short* __restrict__ f2,
                const unsigned short* __restrict__ f3,
                const float* __restrict__ boxes5,
                const int* __restrict__ keys,
                unsigned short* __restrict__ X) {
  int r = blockIdx.x;            // ROI
  int c = threadIdx.x;           // channel
  int b = r >> 10;               // N = 1024
  int key = keys[r];
  const unsigned short* feat; int H;
  switch (key) {
    case 0:  feat = f0; H = 128; break;
    case 1:  feat = f1; H = 64;  break;
    case 2:  feat = f2; H = 32;  break;
    default: feat = f3; H = 16;  break;
  }
  float fsz = (float)H - 1.0f;
  const float* bx = boxes5 + (size_t)r * 5;
  float cx = bx[0], cy = bx[1], w = bx[2], h = bx[3], th = bx[4];
  float ct = cosf(th), st = sinf(th);
  float bw = w * (1.0f / 7.0f), bh = h * (1.0f / 7.0f);
  const unsigned short* fb = feat + (size_t)b * H * H * CCH;
  unsigned short* xr = X + (size_t)r * FIN + (size_t)c * 49;
  for (int oy = 0; oy < 7; ++oy) {
    for (int ox = 0; ox < 7; ++ox) {
      float acc = 0.f;
#pragma unroll
      for (int sy = 0; sy < 2; ++sy) {
#pragma unroll
        for (int sx = 0; sx < 2; ++sx) {
          float offY = (float)oy + ((float)sy + 0.5f) * 0.5f;
          float offX = (float)ox + ((float)sx + 0.5f) * 0.5f;
          float yy = -0.5f * h + offY * bh;
          float xx = -0.5f * w + offX * bw;
          float px = xx * ct - yy * st + cx;
          float py = xx * st + yy * ct + cy;
          px = fminf(fmaxf(px, 0.f), fsz);
          py = fminf(fmaxf(py, 0.f), fsz);
          float fx0 = floorf(px), fy0 = floorf(py);
          float fx1 = fminf(fx0 + 1.f, fsz), fy1 = fminf(fy0 + 1.f, fsz);
          float lx = px - fx0, ly = py - fy0;
          int ix0 = (int)fx0, iy0 = (int)fy0, ix1 = (int)fx1, iy1 = (int)fy1;
          float v00 = bf2f(fb[((size_t)(iy0 * H + ix0)) * CCH + c]);
          float v01 = bf2f(fb[((size_t)(iy0 * H + ix1)) * CCH + c]);
          float v10 = bf2f(fb[((size_t)(iy1 * H + ix0)) * CCH + c]);
          float v11 = bf2f(fb[((size_t)(iy1 * H + ix1)) * CCH + c]);
          acc += v00 * (1.f - ly) * (1.f - lx) + v01 * (1.f - ly) * lx +
                 v10 * ly * (1.f - lx) + v11 * ly * lx;
        }
      }
      xr[oy * 7 + ox] = f2bf(acc * 0.25f);
    }
  }
}

// ---------------- 5. bf16 WMMA GEMM: D = relu(A*B + bias) ----------------
// A: M x K bf16 row-major, B: K x N bf16 row-major, Dst: M x N bf16.
// Block tile 128x128, K-step 32, 8 waves; wave = 2x4 fragments (32x64).
// Ping-pong double buffering with GLOBAL_LOAD_ASYNC_TO_LDS_B128 (ASYNCcnt).
__global__ __launch_bounds__(256)
void k_gemm_bf16(const unsigned short* __restrict__ A,
                 const unsigned short* __restrict__ Bw,
                 const float* __restrict__ bias,
                 unsigned short* __restrict__ Dst,
                 int M, int N, int K, int doRelu) {
  __shared__ __align__(32) unsigned short ldsA[2][128 * 32];
  __shared__ __align__(32) unsigned short ldsB[2][32 * 128];
  int tid = threadIdx.x;
  int lane = tid & 31, wave = tid >> 5;
  int wm = wave & 3, wn = wave >> 2;           // 4x2 wave grid
  int mb = blockIdx.y, nb = blockIdx.x;
  int lm = lane & 15;
  int kg = (lane >> 4) * 8;                    // A-frag K sub-offset per ISA layout

  v8f acc[2][4];
  v8f zero = {};
#pragma unroll
  for (int f = 0; f < 2; ++f)
#pragma unroll
    for (int g = 0; g < 4; ++g) acc[f][g] = zero;

  int arow = tid >> 1, ahalf = (tid & 1) * 16; // 128 rows x 32 cols
  int brow = tid >> 3, bcol = (tid & 7) * 16;  // 32 rows x 128 cols
  const unsigned short* aG = A + (size_t)(mb * 128 + arow) * K + ahalf;
  const unsigned short* bG = Bw + (size_t)brow * N + (size_t)nb * 128 + bcol;

  // per-thread LDS byte offsets (generic ptr low 32 bits == LDS offset)
  unsigned aOff[2], bOff[2];
#pragma unroll
  for (int p = 0; p < 2; ++p) {
    aOff[p] = (unsigned)(unsigned long long)&ldsA[p][arow * 32 + ahalf];
    bOff[p] = (unsigned)(unsigned long long)&ldsB[p][brow * 128 + bcol];
  }

  auto issueTile = [&](int buf, int k0) {
    const unsigned short* ag = aG + k0;
    const unsigned short* bg = bG + (size_t)k0 * N;
    async_ld16(aOff[buf],      (const void*)ag);
    async_ld16(aOff[buf] + 16, (const void*)(ag + 8));
    async_ld16(bOff[buf],      (const void*)bg);
    async_ld16(bOff[buf] + 16, (const void*)(bg + 8));
  };

  issueTile(0, 0);
  int nIter = K >> 5;
  for (int i = 0; i < nIter; ++i) {
    int buf = i & 1;
    int kn = (i + 1) << 5;
    if (kn < K) {
      issueTile(buf ^ 1, kn);
      // 8 outstanding; first 4 (current buffer) complete in order
      asm volatile("s_wait_asynccnt 0x4" ::: "memory");
    } else {
      asm volatile("s_wait_asynccnt 0x0" ::: "memory");
    }
    __syncthreads();

    v16bf afrag[2], bfrag[4];
#pragma unroll
    for (int f = 0; f < 2; ++f) {
      const unsigned short* ar = &ldsA[buf][((wm * 2 + f) * 16 + lm) * 32];
      union { us16 v; us8 h[2]; } u;
      u.h[0] = *(const us8*)(ar + kg);        // K = kg..kg+7
      u.h[1] = *(const us8*)(ar + 16 + kg);   // K = 16+kg..
      afrag[f] = __builtin_bit_cast(v16bf, u.v);
    }
#pragma unroll
    for (int g = 0; g < 4; ++g) {
      us16 bv;
#pragma unroll
      for (int s = 0; s < 16; ++s)
        bv[s] = ldsB[buf][(kg + (s & 7) + (s >> 3) * 16) * 128 + (wn * 4 + g) * 16 + lm];
      bfrag[g] = __builtin_bit_cast(v16bf, bv);
    }
#pragma unroll
    for (int f = 0; f < 2; ++f)
#pragma unroll
      for (int g = 0; g < 4; ++g)
        acc[f][g] = __builtin_amdgcn_wmma_f32_16x16x32_bf16(
            false, afrag[f], false, bfrag[g], (short)0, acc[f][g], false, false);
    __syncthreads();   // protects buffer being refilled next iteration (WAR)
  }

  int rbase = (lane >> 4) * 8;                 // C/D: lanes>=16 hold M=8..15
#pragma unroll
  for (int f = 0; f < 2; ++f) {
#pragma unroll
    for (int g = 0; g < 4; ++g) {
      int gcol = nb * 128 + (wn * 4 + g) * 16 + lm;
      float bv = bias[gcol];
#pragma unroll
      for (int rr = 0; rr < 8; ++rr) {
        int grow = mb * 128 + (wm * 2 + f) * 16 + rbase + rr;
        float v = acc[f][g][rr] + bv;
        if (doRelu) v = fmaxf(v, 0.f);
        Dst[(size_t)grow * N + gcol] = f2bf(v);
      }
    }
  }
}

// ---------------- 6. cls/reg heads: 16 dots of length 1024 per ROI -------
__global__ __launch_bounds__(256)
void k_heads(const unsigned short* __restrict__ H2,
             const float* __restrict__ Wc, const float* __restrict__ bc,
             const float* __restrict__ Wr, const float* __restrict__ br,
             float* __restrict__ clsreg) {
  __shared__ float hrow[HID];
  __shared__ float sdata[256];
  int r = blockIdx.x, t = threadIdx.x;
  const unsigned short* hp = H2 + (size_t)r * HID;
  for (int i = t; i < HID; i += 256) hrow[i] = bf2f(hp[i]);
  __syncthreads();
  int j = t & 15, part = t >> 4;
  float partial = 0.f;
  int kbeg = part * 64;
  if (j < 11) {
    for (int k = kbeg; k < kbeg + 64; ++k) partial += hrow[k] * Wc[k * 11 + j];
  } else {
    int jr = j - 11;
    for (int k = kbeg; k < kbeg + 64; ++k) partial += hrow[k] * Wr[k * 5 + jr];
  }
  sdata[t] = partial;
  __syncthreads();
  if (t < 16) {
    float s = 0.f;
    for (int p = 0; p < 16; ++p) s += sdata[p * 16 + t];
    s += (t < 11) ? bc[t] : br[t - 11];
    clsreg[(size_t)r * 16 + t] = s;
  }
}

// ---------------- 7. softmax + box decode ----------------
__global__ void k_decode(const float* __restrict__ clsreg,
                         const float* __restrict__ boxes5,
                         const int* __restrict__ keys,
                         float* __restrict__ out) {
  int r = blockIdx.x * blockDim.x + threadIdx.x;
  if (r >= NROI) return;
  const float* cr = clsreg + (size_t)r * 16;
  float logits[11];
  float m = -1e30f;
  for (int j = 0; j < 11; ++j) { logits[j] = cr[j]; m = fmaxf(m, logits[j]); }
  float s = 0.f;
  for (int j = 0; j < 11; ++j) { logits[j] = expf(logits[j] - m); s += logits[j]; }
  float inv = 1.f / s;
  float* o = out + (size_t)r * 16;
  for (int j = 0; j < 11; ++j) o[j] = logits[j] * inv;
  const float* bx = boxes5 + (size_t)r * 5;
  float strides[4] = {4.f, 8.f, 16.f, 32.f};
  float sc = strides[keys[r] & 3];
  float e0 = bx[0] * sc, e1 = bx[1] * sc, e2 = bx[2] * sc, e3 = bx[3] * sc, e4 = bx[4];
  float r0 = cr[11], r1 = cr[12], r2 = cr[13], r3 = cr[14], r4 = cr[15];
  const float CLAMP_V = 4.135166556742356f;   // |log(16/1000)|
  float d0 = expf(fminf(fmaxf(r2, -CLAMP_V), CLAMP_V));
  float d1 = expf(fminf(fmaxf(r3, -CLAMP_V), CLAMP_V));
  o[11] = e2 * r0 + e0;
  o[12] = e3 * r1 + e1;
  o[13] = e2 * d0;
  o[14] = e3 * d1;
  o[15] = e4 + r4;
}

// ---------------- launch ----------------
extern "C" void kernel_launch(void* const* d_in, const int* in_sizes, int n_in,
                              void* d_out, int out_size, void* d_ws, size_t ws_size,
                              hipStream_t stream) {
  const float* feat0 = (const float*)d_in[0];
  const float* feat1 = (const float*)d_in[1];
  const float* feat2 = (const float*)d_in[2];
  const float* feat3 = (const float*)d_in[3];
  const float* prop  = (const float*)d_in[4];
  const int*   keys  = (const int*)  d_in[5];
  const float* W1    = (const float*)d_in[6];
  const float* b1    = (const float*)d_in[7];
  const float* W2    = (const float*)d_in[8];
  const float* b2    = (const float*)d_in[9];
  const float* Wc    = (const float*)d_in[10];
  const float* bc    = (const float*)d_in[11];
  const float* Wr    = (const float*)d_in[12];
  const float* br    = (const float*)d_in[13];
  float* out = (float*)d_out;

  // workspace layout (256B aligned)
  char* base = (char*)d_ws;
  size_t off = 0;
  auto take = [&](size_t bytes) {
    char* p = base + off;
    off = (off + bytes + 255) & ~(size_t)255;
    return p;
  };
  float*          boxes5 = (float*)         take((size_t)NROI * 5 * 4);
  unsigned short* nhwc0  = (unsigned short*)take((size_t)NB * 128 * 128 * CCH * 2);
  unsigned short* nhwc1  = (unsigned short*)take((size_t)NB * 64 * 64 * CCH * 2);
  unsigned short* nhwc2  = (unsigned short*)take((size_t)NB * 32 * 32 * CCH * 2);
  unsigned short* nhwc3  = (unsigned short*)take((size_t)NB * 16 * 16 * CCH * 2);
  unsigned short* Xb     = (unsigned short*)take((size_t)NROI * FIN * 2);
  unsigned short* W1b    = (unsigned short*)take((size_t)FIN * HID * 2);
  unsigned short* W2b    = (unsigned short*)take((size_t)HID * HID * 2);
  unsigned short* H1b    = (unsigned short*)take((size_t)NROI * HID * 2);
  unsigned short* H2b    = (unsigned short*)take((size_t)NROI * HID * 2);
  float*          clsreg = (float*)         take((size_t)NROI * 16 * 4);
  (void)ws_size; (void)in_sizes; (void)n_in; (void)out_size;

  // weight conversion
  {
    int n1 = FIN * HID;
    k_cvt_bf16<<<(n1 + 255) / 256, 256, 0, stream>>>(W1, W1b, n1);
    int n2 = HID * HID;
    k_cvt_bf16<<<(n2 + 255) / 256, 256, 0, stream>>>(W2, W2b, n2);
  }
  // feature layout transform
  {
    int n0 = NB * CCH * 128 * 128;
    k_nchw_to_nhwc<<<(n0 + 255) / 256, 256, 0, stream>>>(feat0, nhwc0, 128);
    int n1 = NB * CCH * 64 * 64;
    k_nchw_to_nhwc<<<(n1 + 255) / 256, 256, 0, stream>>>(feat1, nhwc1, 64);
    int n2 = NB * CCH * 32 * 32;
    k_nchw_to_nhwc<<<(n2 + 255) / 256, 256, 0, stream>>>(feat2, nhwc2, 32);
    int n3 = NB * CCH * 16 * 16;
    k_nchw_to_nhwc<<<(n3 + 255) / 256, 256, 0, stream>>>(feat3, nhwc3, 16);
  }
  // box encode
  k_encode<<<(NROI + 255) / 256, 256, 0, stream>>>(prop, boxes5);
  // ROI align
  k_roialign<<<NROI, 256, 0, stream>>>(nhwc0, nhwc1, nhwc2, nhwc3, boxes5, keys, Xb);
  // FC1: (2048 x 12544) @ (12544 x 1024)
  k_gemm_bf16<<<dim3(HID / 128, NROI / 128), 256, 0, stream>>>(
      Xb, W1b, b1, H1b, NROI, HID, FIN, 1);
  // FC2: (2048 x 1024) @ (1024 x 1024)
  k_gemm_bf16<<<dim3(HID / 128, NROI / 128), 256, 0, stream>>>(
      H1b, W2b, b2, H2b, NROI, HID, HID, 1);
  // heads
  k_heads<<<NROI, 256, 0, stream>>>(H2b, Wc, bc, Wr, br, clsreg);
  // softmax + decode
  k_decode<<<(NROI + 255) / 256, 256, 0, stream>>>(clsreg, boxes5, keys, out);
}
